// MSA_6262062317876
// MI455X (gfx1250) — matrix operand order repaired and checked by reading, not compiled, not run on hardware
//
#include <hip/hip_runtime.h>

typedef __attribute__((ext_vector_type(16))) _Float16 v16h;
typedef __attribute__((ext_vector_type(8)))  float    v8f;

// ---------------------------------------------------------------------------
// Async global->LDS copy (gfx1250 ASYNCcnt path), with safe fallback.
// Builtin signature (from hipcc diagnostic): arg0 = int4(vector_size 16) in
// AS1 (__device__), arg1 = LDS pointer (AS3), arg2/3 = imm offset / cpol.
// ---------------------------------------------------------------------------
#if __has_builtin(__builtin_amdgcn_global_load_async_to_lds_b128) && \
    __has_builtin(__builtin_amdgcn_s_wait_asynccnt)
#define HAVE_ASYNC_LDS 1
#else
#define HAVE_ASYNC_LDS 0
#endif

typedef int v4i_vs __attribute__((vector_size(16)));
typedef __attribute__((address_space(1))) v4i_vs* gptr128_t;
typedef __attribute__((address_space(3))) v4i_vs* lptr128_t;

__device__ __forceinline__ void async_copy16(const _Float16* g, _Float16* l) {
#if HAVE_ASYNC_LDS
  __builtin_amdgcn_global_load_async_to_lds_b128(
      (gptr128_t)(const void*)g, (lptr128_t)(void*)l, 0, 0);
#else
  *(uint4*)l = *(const uint4*)g;
#endif
}
__device__ __forceinline__ void async_join() {
#if HAVE_ASYNC_LDS
  __builtin_amdgcn_s_wait_asynccnt(0);
#endif
}

// ---------------------------------------------------------------------------
// Fragment loaders (CDNA5 WMMA 16x16x32 f16 layouts, wave32)
// A (16Mx32K): lane M = lane%16. lanes<16: halfs0-7 -> K0-7, halfs8-15 -> K16-23
//              lanes>=16: halfs0-7 -> K8-15, halfs8-15 -> K24-31
// B (32Kx16N): lane N = lane%16. same K order as A (halfs0-7 -> K 8*(lane/16)+h, halfs8-15 -> K 16+8*(lane/16)+h-8)
// C/D (16x16 f32): vgpr r -> M = r + 8*(lane/16), N = lane%16
// ---------------------------------------------------------------------------
__device__ __forceinline__ v16h load_a_frag(const _Float16* base, int stride_h,
                                            int row, int k0, int lane) {
  union { v16h v; unsigned u[8]; } f;
  const _Float16* rp = base + row * stride_h;
  const int hi = (lane >> 4) & 1;
#pragma unroll
  for (int p = 0; p < 8; ++p) {
    int k = 2 * p + ((p < 4) ? 0 : 8) + hi * 8 + k0;
    f.u[p] = *(const unsigned*)(rp + k);
  }
  return f.v;
}

__device__ __forceinline__ v16h load_b_frag(const _Float16* base, int stride_h,
                                            int nrow, int k0, int lane) {
  union { v16h v; unsigned u[8]; } f;
  const _Float16* rp = base + nrow * stride_h + k0 + ((lane >> 4) & 1) * 8;
#pragma unroll
  for (int p = 0; p < 8; ++p) f.u[p] = *(const unsigned*)(rp + 2 * p + ((p < 4) ? 0 : 8));
  return f.v;
}

__device__ __forceinline__ v8f wmma_f16(v16h a, v16h b, v8f c) {
  return __builtin_amdgcn_wmma_f32_16x16x32_f16(false, a, false, b,
                                                (short)0, c, false, false);
}

// ---------------------------------------------------------------------------
// Kernel 1: weight prep.  Writes TRANSPOSED f16 weights:
//   W16T [3072 n][1024 k]  = [Wq*0.125 | Wk | Wv]^T   (scale folded into Wq)
//   Wo16T[1024 n][1024 k]  = Wo^T
// LDS-tiled 32x32 transpose, coalesced both directions.
// grid (32, 128), block (32, 8)
// ---------------------------------------------------------------------------
__global__ __launch_bounds__(256) void prep_weights(
    const float* __restrict__ Wq, const float* __restrict__ Wk,
    const float* __restrict__ Wv, const float* __restrict__ Wo,
    _Float16* __restrict__ W16T, _Float16* __restrict__ Wo16T) {
  __shared__ _Float16 tile[32][33];
  const int tx = threadIdx.x, ty = threadIdx.y;
  const int kt = blockIdx.x * 32;
  const bool isWo = blockIdx.y >= 96;
  const int nt = (isWo ? (blockIdx.y - 96) : blockIdx.y) * 32;
#pragma unroll
  for (int i = 0; i < 4; ++i) {
    int k = kt + ty + i * 8;
    int n = nt + tx;
    float v;
    if (isWo)            v = Wo[k * 1024 + n];
    else if (n < 1024)   v = Wq[k * 1024 + n] * 0.125f;
    else if (n < 2048)   v = Wk[k * 1024 + (n - 1024)];
    else                 v = Wv[k * 1024 + (n - 2048)];
    tile[ty + i * 8][tx] = (_Float16)v;
  }
  __syncthreads();
#pragma unroll
  for (int i = 0; i < 4; ++i) {
    int n = nt + ty + i * 8;
    int k = kt + tx;
    _Float16 h = tile[tx][ty + i * 8];
    if (isWo) Wo16T[(size_t)n * 1024 + k] = h;
    else      W16T[(size_t)n * 1024 + k]  = h;
  }
}

// ---------------------------------------------------------------------------
// Kernel 3: V transpose: v16 [bh][n][64] -> v16t [bh][64][n]
// grid (64, 2, 64), block (32, 8)
// ---------------------------------------------------------------------------
__global__ __launch_bounds__(256) void transpose_v(
    const _Float16* __restrict__ v16, _Float16* __restrict__ v16t) {
  __shared__ _Float16 tile[32][33];
  const int tx = threadIdx.x, ty = threadIdx.y;
  const int ntile = blockIdx.x * 32;
  const int dtile = blockIdx.y * 32;
  const size_t base  = (size_t)blockIdx.z * 2048 * 64;
#pragma unroll
  for (int i = 0; i < 4; ++i) {
    int n = ntile + ty + i * 8;
    int d = dtile + tx;
    tile[ty + i * 8][tx] = v16[base + (size_t)n * 64 + d];
  }
  __syncthreads();
#pragma unroll
  for (int i = 0; i < 4; ++i) {
    int d = dtile + ty + i * 8;
    int n = ntile + tx;
    v16t[base + (size_t)d * 2048 + n] = tile[tx][ty + i * 8];
  }
}

// ---------------------------------------------------------------------------
// Kernel 2/5: 128x128 block-tile WMMA GEMM, double-buffered LDS, async
// global->LDS staging.  B is pre-transposed: BwT [n][k], row stride = K.
// 256 threads = 8 waves (2Mx4N), each wave computes 64x32 (4x2 WMMA tiles).
// MODE 0: A = f32 (x), epilogue scatters Q/K/V f16 into [B,H,N,64]
// MODE 1: A = f16 (attn out), epilogue adds bias, writes f32 out
// ---------------------------------------------------------------------------
template <int MODE>
__global__ __launch_bounds__(256) void gemm128(
    const void* __restrict__ Aptr, const _Float16* __restrict__ BwT,
    int K, int Nw,
    _Float16* __restrict__ q16, _Float16* __restrict__ k16,
    _Float16* __restrict__ v16,
    const float* __restrict__ bias, float* __restrict__ outf) {
  __shared__ _Float16 Alds[2][128][40];   // [buf][m][k]
  __shared__ _Float16 Blds[2][128][40];   // [buf][n][k]

  const int tid  = threadIdx.x;
  const int lane = tid & 31;
  const int wave = tid >> 5;
  const int wm   = wave >> 2;          // 0..1
  const int wn   = wave & 3;           // 0..3
  const int m0   = blockIdx.x * 128;
  const int n0   = blockIdx.y * 128;

  auto stageA = [&](int buf, int kk) {
    if (MODE == 0) {
      const float* A = (const float*)Aptr;
#pragma unroll
      for (int s = 0; s < 4; ++s) {
        int r = s * 32 + (tid >> 3);
        int c = (tid & 7) * 4;
        float4 x4 = *(const float4*)(A + (size_t)(m0 + r) * K + kk + c);
        union { uint2 u2; _Float16 h[4]; } o;
        o.h[0] = (_Float16)x4.x; o.h[1] = (_Float16)x4.y;
        o.h[2] = (_Float16)x4.z; o.h[3] = (_Float16)x4.w;
        *(uint2*)&Alds[buf][r][c] = o.u2;
      }
    } else {
      const _Float16* A = (const _Float16*)Aptr;
#pragma unroll
      for (int s = 0; s < 2; ++s) {
        int idx = s * 256 + tid;        // 0..511 (8-half granules)
        int r = idx >> 2;               // 0..127
        int c = (idx & 3) * 8;          // 0,8,16,24
        async_copy16(A + (size_t)(m0 + r) * K + kk + c, &Alds[buf][r][c]);
      }
    }
  };
  auto stageB = [&](int buf, int kk) {
#pragma unroll
    for (int s = 0; s < 2; ++s) {
      int idx = s * 256 + tid;
      int r = idx >> 2;                 // n row 0..127
      int c = (idx & 3) * 8;
      async_copy16(BwT + (size_t)(n0 + r) * K + kk + c, &Blds[buf][r][c]);
    }
  };

  v8f acc[4][2];
#pragma unroll
  for (int i = 0; i < 4; ++i)
#pragma unroll
    for (int j = 0; j < 2; ++j) acc[i][j] = (v8f)(0.0f);

  stageA(0, 0);
  stageB(0, 0);
  async_join();
  __syncthreads();

  const int nsteps = K >> 5;
  for (int s = 0; s < nsteps; ++s) {
    const int cur = s & 1;
    // prefetch two steps ahead (global_prefetch_b8)
    if (s + 2 < nsteps) {
      const int kk2 = (s + 2) << 5;
      if (MODE == 0)
        __builtin_prefetch((const float*)Aptr + (size_t)(m0 + (tid >> 1)) * K + kk2 + (tid & 1) * 16, 0, 3);
      else
        __builtin_prefetch((const _Float16*)Aptr + (size_t)(m0 + (tid >> 1)) * K + kk2 + (tid & 1) * 16, 0, 3);
      __builtin_prefetch(BwT + (size_t)(n0 + (tid >> 1)) * K + kk2 + (tid & 1) * 16, 0, 3);
    }
    // stage next tile while computing current (ping-pong)
    if (s + 1 < nsteps) {
      stageA(cur ^ 1, (s + 1) << 5);
      stageB(cur ^ 1, (s + 1) << 5);
    }

    v16h af[4], bf[2];
#pragma unroll
    for (int i = 0; i < 4; ++i)
      af[i] = load_a_frag(&Alds[cur][0][0], 40, wm * 64 + i * 16 + (lane & 15), 0, lane);
#pragma unroll
    for (int j = 0; j < 2; ++j)
      bf[j] = load_b_frag(&Blds[cur][0][0], 40, wn * 32 + j * 16 + (lane & 15), 0, lane);
#pragma unroll
    for (int i = 0; i < 4; ++i)
#pragma unroll
      for (int j = 0; j < 2; ++j) acc[i][j] = wmma_f16(af[i], bf[j], acc[i][j]);
    async_join();
    __syncthreads();
  }

  // ---- epilogue ----
#pragma unroll
  for (int i = 0; i < 4; ++i) {
#pragma unroll
    for (int j = 0; j < 2; ++j) {
      const int nn = n0 + wn * 32 + j * 16 + (lane & 15);
      if (MODE == 0) {
        const int mat = nn >> 10;            // 0=q 1=k 2=v
        const int col = nn & 1023;           // h*64 + d
        const int hh  = col >> 6, dd = col & 63;
        _Float16* dst = (mat == 0) ? q16 : ((mat == 1) ? k16 : v16);
#pragma unroll
        for (int r = 0; r < 8; ++r) {
          int mm = m0 + wm * 64 + i * 16 + r + ((lane >> 4) << 3);
          int b = mm >> 11, n = mm & 2047;   // row = b*2048 + n
          size_t off = (((size_t)(b * 16 + hh) * 2048) + n) * 64 + dd;
          dst[off] = (_Float16)acc[i][j][r];
        }
      } else {
        const float bv = bias[nn];
#pragma unroll
        for (int r = 0; r < 8; ++r) {
          int mm = m0 + wm * 64 + i * 16 + r + ((lane >> 4) << 3);
          outf[(size_t)mm * Nw + nn] = acc[i][j][r] + bv;
        }
      }
    }
  }
}

// ---------------------------------------------------------------------------
// Kernel 4: flash attention.  grid = (N/128, B*H), block = 256 (8 waves).
// Each wave owns 16 query rows x full Dh=64.  Key blocks of 64.
// V is pre-transposed in global memory -> all staging is aligned b128 async.
// Row-sum of P via WMMA against ones matrix; only row-max uses shuffles.
// Output written to [B,N,H*64] f16.
// ---------------------------------------------------------------------------
__global__ __launch_bounds__(256) void flash_attn(
    const _Float16* __restrict__ q16, const _Float16* __restrict__ k16,
    const _Float16* __restrict__ v16t, _Float16* __restrict__ o16) {
  __shared__ _Float16 Qlds[128][72];       // [qrow][d]
  __shared__ _Float16 Klds[64][72];        // [key][d]
  __shared__ _Float16 Vlds[64][72];        // [d][key]
  __shared__ _Float16 Plds[8][16][72];     // per-wave P tile [m][key 0..63]

  const int tid  = threadIdx.x;
  const int lane = tid & 31;
  const int wave = tid >> 5;               // 0..7
  const int bh   = blockIdx.y;             // b*16 + h
  const int q0   = blockIdx.x * 128;
  const size_t base = (size_t)bh * 2048 * 64;   // same total for v16t

  // stage Q block (128x64): 8192 halfs, b128 async, 4 iters
#pragma unroll
  for (int s = 0; s < 4; ++s) {
    int idx = s * 256 + tid;
    int r = idx >> 3, c = (idx & 7) * 8;
    async_copy16(q16 + base + (size_t)(q0 + r) * 64 + c, &Qlds[r][c]);
  }
  async_join();
  __syncthreads();

  v16h qf[2];
  qf[0] = load_a_frag(&Qlds[0][0], 72, wave * 16 + (lane & 15), 0, lane);
  qf[1] = load_a_frag(&Qlds[0][0], 72, wave * 16 + (lane & 15), 32, lane);

  v16h ones;
#pragma unroll
  for (int i = 0; i < 16; ++i) ones[i] = (_Float16)1.0f;

  float mrow[8], lrow[8];
  v8f ox[4];
#pragma unroll
  for (int r = 0; r < 8; ++r) { mrow[r] = -1e30f; lrow[r] = 0.0f; }
#pragma unroll
  for (int j = 0; j < 4; ++j) ox[j] = (v8f)(0.0f);

  for (int kb = 0; kb < 2048; kb += 64) {
    // prefetch next K/V block
    if (kb + 64 < 2048) {
      int r = tid >> 2, c = (tid & 3) * 16;
      __builtin_prefetch(k16 + base + (size_t)(kb + 64 + r) * 64 + c, 0, 3);
      __builtin_prefetch(v16t + base + (size_t)(tid >> 2) * 2048 + kb + 64, 0, 3);
    }
    // stage K (64 keys x 64 d) and V^T (64 d x 64 keys): b128 async, 2 iters each
#pragma unroll
    for (int s = 0; s < 2; ++s) {
      int idx = s * 256 + tid;
      int r = idx >> 3, c = (idx & 7) * 8;
      async_copy16(k16 + base + (size_t)(kb + r) * 64 + c, &Klds[r][c]);
    }
#pragma unroll
    for (int s = 0; s < 2; ++s) {
      int idx = s * 256 + tid;
      int r = idx >> 3, c = (idx & 7) * 8;   // r = d, c = key offset
      async_copy16(v16t + base + (size_t)r * 2048 + kb + c, &Vlds[r][c]);
    }
    async_join();
    __syncthreads();

    // S = Q K^T : four 16x16 key tiles, K-dim = 64 (two WMMAs each)
    v8f sacc[4];
#pragma unroll
    for (int t = 0; t < 4; ++t) sacc[t] = (v8f)(0.0f);
#pragma unroll
    for (int dk = 0; dk < 2; ++dk) {
#pragma unroll
      for (int t = 0; t < 4; ++t) {
        v16h bk = load_b_frag(&Klds[0][0], 72, t * 16 + (lane & 15), dk * 32, lane);
        sacc[t] = wmma_f16(qf[dk], bk, sacc[t]);
      }
    }

    // online softmax: row-max via shuffles; P written to LDS as f16
#pragma unroll
    for (int r = 0; r < 8; ++r) {
      float t01 = fmaxf(sacc[0][r], sacc[1][r]);
      float t23 = fmaxf(sacc[2][r], sacc[3][r]);
      float t = fmaxf(t01, t23);
      t = fmaxf(t, __shfl_xor(t, 1, 32));
      t = fmaxf(t, __shfl_xor(t, 2, 32));
      t = fmaxf(t, __shfl_xor(t, 4, 32));
      t = fmaxf(t, __shfl_xor(t, 8, 32));
      float mnew = fmaxf(mrow[r], t);
      float corr = __expf(mrow[r] - mnew);
      mrow[r] = mnew;
      lrow[r] *= corr;
#pragma unroll
      for (int j = 0; j < 4; ++j) ox[j][r] *= corr;
      int mloc = r + ((lane >> 4) << 3);
#pragma unroll
      for (int tt = 0; tt < 4; ++tt) {
        float p = __expf(sacc[tt][r] - mnew);
        Plds[wave][mloc][tt * 16 + (lane & 15)] = (_Float16)p;
      }
    }

    // P fragments (wave-local LDS round trip; DS ops in-order per wave)
    v16h pf0 = load_a_frag(&Plds[wave][0][0], 72, (lane & 15), 0, lane);
    v16h pf1 = load_a_frag(&Plds[wave][0][0], 72, (lane & 15), 32, lane);

    // row-sum of P via WMMA with ones matrix
    v8f lblk = (v8f)(0.0f);
    lblk = wmma_f16(pf0, ones, lblk);
    lblk = wmma_f16(pf1, ones, lblk);

    // O += P V
#pragma unroll
    for (int j = 0; j < 4; ++j) {
      v16h bv0 = load_b_frag(&Vlds[0][0], 72, j * 16 + (lane & 15), 0, lane);
      v16h bv1 = load_b_frag(&Vlds[0][0], 72, j * 16 + (lane & 15), 32, lane);
      ox[j] = wmma_f16(pf0, bv0, ox[j]);
      ox[j] = wmma_f16(pf1, bv1, ox[j]);
    }
#pragma unroll
    for (int r = 0; r < 8; ++r) lrow[r] += lblk[r];
    __syncthreads();
  }

  // epilogue: normalize and write to [B, N, H*64] f16
  const int b = bh >> 4, h = bh & 15;
#pragma unroll
  for (int j = 0; j < 4; ++j) {
    int dd = j * 16 + (lane & 15);
#pragma unroll
    for (int r = 0; r < 8; ++r) {
      int mm = r + ((lane >> 4) << 3);
      int n  = q0 + wave * 16 + mm;
      float val = ox[j][r] / lrow[r];
      o16[((size_t)(b * 2048 + n) * 1024) + h * 64 + dd] = (_Float16)val;
    }
  }
}

// ---------------------------------------------------------------------------
// launch
// ---------------------------------------------------------------------------
extern "C" void kernel_launch(void* const* d_in, const int* in_sizes, int n_in,
                              void* d_out, int out_size, void* d_ws, size_t ws_size,
                              hipStream_t stream) {
  const float* x  = (const float*)d_in[0];
  const float* Wq = (const float*)d_in[1];
  const float* Wk = (const float*)d_in[2];
  const float* Wv = (const float*)d_in[3];
  const float* Wo = (const float*)d_in[4];
  const float* bo = (const float*)d_in[5];
  float* out = (float*)d_out;

  char* ws = (char*)d_ws;
  _Float16* W16T  = (_Float16*)(ws);                   //  6 MB: [3072 n][1024 k]
  _Float16* Wo16T = (_Float16*)(ws + 6291456);         //  2 MB: [1024 n][1024 k]
  _Float16* q16   = (_Float16*)(ws + 8388608);         // 16 MB: [B,H,N,64]
  _Float16* k16   = (_Float16*)(ws + 25165824);        // 16 MB
  _Float16* v16   = (_Float16*)(ws + 41943040);        // 16 MB
  _Float16* v16t  = (_Float16*)(ws + 58720256);        // 16 MB: [B,H,64,N]
  _Float16* o16   = (_Float16*)(ws + 75497472);        // 16 MB: [B,N,1024]

  prep_weights<<<dim3(32, 128), dim3(32, 8), 0, stream>>>(
      Wq, Wk, Wv, Wo, W16T, Wo16T);

  // QKV projection: [8192 x 1024] x [1024 x 3072]
  gemm128<0><<<dim3(64, 24), 256, 0, stream>>>(
      (const void*)x, W16T, 1024, 3072, q16, k16, v16, nullptr, nullptr);

  // V transpose for attention B-fragments
  transpose_v<<<dim3(64, 2, 64), dim3(32, 8), 0, stream>>>(v16, v16t);

  // attention
  flash_attn<<<dim3(16, 64), 256, 0, stream>>>(q16, k16, v16t, o16);

  // output projection: [8192 x 1024] x [1024 x 1024] + bias
  gemm128<1><<<dim3(64, 8), 256, 0, stream>>>(
      (const void*)o16, Wo16T, 1024, 1024, nullptr, nullptr, nullptr, bo, out);
}
